// RnnTfModel_85057532330354
// MI455X (gfx1250) — matrix operand order repaired and checked by reading, not compiled, not run on hardware
//
#include <hip/hip_runtime.h>
#include <hip/hip_bf16.h>
#include <math.h>

typedef __attribute__((ext_vector_type(16))) _Float16 v16h;
typedef __attribute__((ext_vector_type(8)))  _Float16 v8h;
typedef __attribute__((ext_vector_type(8)))  float    v8f;

#define B_TOT 512
#define T_LEN 1024
#define DH    64
#define NZ    256   // 4*DH gate width
#define MT    16    // batch rows per workgroup
#define XPAD  72    // LDS row stride in halves (144B, 16B-divisible)

// ---- fast activations: gfx1250 has v_tanh_f32 ----
#if __has_builtin(__builtin_amdgcn_tanhf)
__device__ __forceinline__ float fast_tanh(float x) { return __builtin_amdgcn_tanhf(x); }
__device__ __forceinline__ float fast_sigmoid(float x) {
    return fmaf(0.5f, __builtin_amdgcn_tanhf(0.5f * x), 0.5f);
}
#else
__device__ __forceinline__ float fast_tanh(float x) {
    // tanh(x) = 1 - 2/(exp(2x)+1), v_exp + v_rcp, no divides/branches
    return 1.0f - 2.0f * __builtin_amdgcn_rcpf(__expf(2.0f * x) + 1.0f);
}
__device__ __forceinline__ float fast_sigmoid(float x) {
    return __builtin_amdgcn_rcpf(1.0f + __expf(-x));
}
#endif

// A-matrix 16x32 f16 fragment from LDS tile buf[16][XPAD], K-tile at k0.
// ISA 7.12.2: lane (m=lane&15, half=lane>>4); v16h elem j -> K = k0 + 8*half + j (j<8)
//                                                         K = k0 + 16 + 8*half + (j-8) (j>=8)
__device__ __forceinline__ v16h load_a_frag(const _Float16* buf, int lane, int k0) {
    const int m = lane & 15, half = lane >> 4;
    const v8h lo = *(const v8h*)(buf + m * XPAD + k0 + 8 * half);
    const v8h hi = *(const v8h*)(buf + m * XPAD + k0 + 16 + 8 * half);
    v16h r;
    #pragma unroll
    for (int j = 0; j < 8; ++j) { r[j] = lo[j]; r[j + 8] = hi[j]; }
    return r;
}

__global__ __launch_bounds__(128) void lstm_layer_kernel(
    const float* __restrict__ W, const float* __restrict__ U,
    const float* __restrict__ bias,
    const int* __restrict__ code, const float* __restrict__ emb,   // layer 1 inputs
    const _Float16* __restrict__ xseq,                             // layer 2 input [T][B][64] f16
    _Float16* __restrict__ hseq_out,                               // layer 1 output sequence
    float* __restrict__ h2_out,                                    // layer 2 final hidden
    int layer)
{
    __shared__ __align__(16) _Float16 xbuf[MT * XPAD];
    __shared__ __align__(16) _Float16 hbuf[MT * XPAD];
    __shared__ __align__(32) float act[MT * NZ];   // post-activation gates i|f|g|o

    const int tid   = threadIdx.x;
    const int lane  = tid & 31;
    const int wave  = tid >> 5;       // wave w == gate w (i,f,g,o)
    const int bbase = blockIdx.x * MT;
    const int goff  = wave * 64;
    const int m     = lane & 15, half = lane >> 4;

    // Branchless per-wave activation: act = amul * tanh(ascale*x) + aoff
    //   sigmoid gates (i,f,o): 0.5*tanh(0.5x)+0.5 ; tanh gate (g): 1*tanh(x)+0
    const float ascale = (wave == 2) ? 1.0f : 0.5f;
    const float amul   = (wave == 2) ? 1.0f : 0.5f;
    const float aoff   = (wave == 2) ? 0.0f : 0.5f;

    // ---- preload weight B-fragments into VGPRs (resident across all T steps) ----
    // B 32x16 f16 layout: lane -> N = goff+tn*16+(lane&15); elem j -> K = kt*32 + 8*half + j (j<8), kt*32 + 16 + 8*half + (j-8) (j>=8)
    v16h wfrag[4][2], ufrag[4][2];
    #pragma unroll
    for (int tn = 0; tn < 4; ++tn) {
        const int ncol = goff + tn * 16 + m;
        #pragma unroll
        for (int kt = 0; kt < 2; ++kt) {
            const int kb = kt * 32 + 8 * half;
            v16h wf, uf;
            #pragma unroll
            for (int j = 0; j < 16; ++j) {
                wf[j] = (_Float16)W[(size_t)(kb + (j < 8 ? j : j + 8)) * NZ + ncol];
                uf[j] = (_Float16)U[(size_t)(kb + (j < 8 ? j : j + 8)) * NZ + ncol];
            }
            wfrag[tn][kt] = wf; ufrag[tn][kt] = uf;
        }
    }

    // per-thread 8-element slice of the [16][64] state tile
    const int erow = tid >> 3;        // 0..15
    const int ecol = (tid & 7) * 8;   // 0..56 step 8
    float cst[8];
    #pragma unroll
    for (int e = 0; e < 8; ++e) cst[e] = 0.f;

    float bv[4];
    #pragma unroll
    for (int tn = 0; tn < 4; ++tn) bv[tn] = bias[goff + tn * 16 + m];

    // zero h state, stage x_0
    for (int i = tid; i < MT * XPAD; i += 128) hbuf[i] = (_Float16)0.f;
    if (layer == 1) {
        const int cidx = code[(size_t)(bbase + erow) * T_LEN + 0];
        const float* er = emb + (size_t)cidx * DH + ecol;
        v8h xv;
        #pragma unroll
        for (int j = 0; j < 8; ++j) xv[j] = (_Float16)er[j];
        *(v8h*)(xbuf + erow * XPAD + ecol) = xv;
    } else {
        *(v8h*)(xbuf + erow * XPAD + ecol) =
            *(const v8h*)(xseq + ((size_t)0 * B_TOT + bbase + erow) * DH + ecol);
    }
    __syncthreads();

    for (int t = 0; t < T_LEN; ++t) {
        // ---- phase A: z[:, gate w] = x_t @ W + h @ U + b via WMMA, store activated gates ----
        const v16h xa0 = load_a_frag(xbuf, lane, 0);
        const v16h xa1 = load_a_frag(xbuf, lane, 32);
        const v16h ha0 = load_a_frag(hbuf, lane, 0);
        const v16h ha1 = load_a_frag(hbuf, lane, 32);

        #pragma unroll
        for (int tn = 0; tn < 4; ++tn) {
            const float b0 = bv[tn];
            v8f acc = {b0, b0, b0, b0, b0, b0, b0, b0};
            acc = __builtin_amdgcn_wmma_f32_16x16x32_f16(false, xa0, false, wfrag[tn][0], (short)0, acc, false, false);
            acc = __builtin_amdgcn_wmma_f32_16x16x32_f16(false, xa1, false, wfrag[tn][1], (short)0, acc, false, false);
            acc = __builtin_amdgcn_wmma_f32_16x16x32_f16(false, ha0, false, ufrag[tn][0], (short)0, acc, false, false);
            acc = __builtin_amdgcn_wmma_f32_16x16x32_f16(false, ha1, false, ufrag[tn][1], (short)0, acc, false, false);
            // C/D layout: elem r -> row (r + 8*half), col = goff + tn*16 + m
            const int col = goff + tn * 16 + m;
            #pragma unroll
            for (int r = 0; r < 8; ++r)
                act[(r + 8 * half) * NZ + col] = fmaf(amul, fast_tanh(ascale * acc[r]), aoff);
        }
        __syncthreads();

        // ---- phase B: gate fusion + state update, stage x_{t+1} ----
        const float* arow = act + erow * NZ;
        const v8f iv = *(const v8f*)(arow + ecol);
        const v8f fv = *(const v8f*)(arow + 64  + ecol);
        const v8f gv = *(const v8f*)(arow + 128 + ecol);
        const v8f ov = *(const v8f*)(arow + 192 + ecol);
        v8h hv8;
        #pragma unroll
        for (int e = 0; e < 8; ++e) {
            const float cv = fv[e] * cst[e] + iv[e] * gv[e];
            cst[e] = cv;
            const float hv = ov[e] * fast_tanh(cv);
            hv8[e] = (_Float16)hv;
            if (layer == 2 && t == T_LEN - 1)
                h2_out[(size_t)(bbase + erow) * DH + ecol + e] = hv;
        }
        *(v8h*)(hbuf + erow * XPAD + ecol) = hv8;
        if (layer == 1)
            *(v8h*)(hseq_out + ((size_t)t * B_TOT + bbase + erow) * DH + ecol) = hv8;

        if (t + 1 < T_LEN) {
            if (layer == 1) {
                const int cidx = code[(size_t)(bbase + erow) * T_LEN + (t + 1)];
                const float* er = emb + (size_t)cidx * DH + ecol;
                v8h xv;
                #pragma unroll
                for (int j = 0; j < 8; ++j) xv[j] = (_Float16)er[j];
                *(v8h*)(xbuf + erow * XPAD + ecol) = xv;
            } else {
                *(v8h*)(xbuf + erow * XPAD + ecol) =
                    *(const v8h*)(xseq + ((size_t)(t + 1) * B_TOT + bbase + erow) * DH + ecol);
                if (t + 8 < T_LEN)  // prefetch a few steps ahead into cache
                    __builtin_prefetch(xseq + ((size_t)(t + 8) * B_TOT + bbase + erow) * DH + ecol, 0, 1);
            }
        }
        __syncthreads();
    }
}

__global__ __launch_bounds__(128) void head_kernel(
    const float* __restrict__ h2,  const float* __restrict__ aux,
    const float* __restrict__ Wlm, const float* __restrict__ blm,
    const float* __restrict__ gam, const float* __restrict__ bet,
    const float* __restrict__ mea, const float* __restrict__ var,
    const float* __restrict__ W3,  const float* __restrict__ b3,
    const float* __restrict__ W4,  const float* __restrict__ b4,
    float* __restrict__ out)
{
    const int b = blockIdx.x * blockDim.x + threadIdx.x;
    if (b >= B_TOT) return;

    float h[64];
    #pragma unroll
    for (int d = 0; d < 64; ++d) h[d] = h2[(size_t)b * 64 + d];

    // language-model head: sigmoid(h2 @ Wlm + blm) -> second output block
    #pragma unroll
    for (int k = 0; k < 2; ++k) {
        float s = blm[k];
        #pragma unroll
        for (int d = 0; d < 64; ++d) s += h[d] * Wlm[d * 2 + k];
        out[B_TOT * 2 + b * 2 + k] = fast_sigmoid(s);
    }

    // z = BN(concat(aux, h2))
    float z[66];
    z[0] = aux[b * 2 + 0];
    z[1] = aux[b * 2 + 1];
    #pragma unroll
    for (int d = 0; d < 64; ++d) z[2 + d] = h[d];
    #pragma unroll
    for (int j = 0; j < 66; ++j)
        z[j] = gam[j] * (z[j] - mea[j]) * rsqrtf(var[j] + 1e-3f) + bet[j];

    // relu(z @ W3 + b3) then logits = y @ W4 + b4, softmax
    float l0 = b4[0], l1 = b4[1];
    #pragma unroll 4
    for (int k = 0; k < 32; ++k) {
        float s = b3[k];
        #pragma unroll
        for (int j = 0; j < 66; ++j) s += z[j] * W3[j * 32 + k];
        s = fmaxf(s, 0.f);
        l0 += s * W4[k * 2 + 0];
        l1 += s * W4[k * 2 + 1];
    }
    const float mx = fmaxf(l0, l1);
    const float e0 = __expf(l0 - mx), e1 = __expf(l1 - mx);
    const float inv = __builtin_amdgcn_rcpf(e0 + e1);
    out[b * 2 + 0] = e0 * inv;
    out[b * 2 + 1] = e1 * inv;
}

extern "C" void kernel_launch(void* const* d_in, const int* in_sizes, int n_in,
                              void* d_out, int out_size, void* d_ws, size_t ws_size,
                              hipStream_t stream) {
    const float* aux  = (const float*)d_in[0];
    const int*   code = (const int*)  d_in[1];
    const float* emb  = (const float*)d_in[2];
    const float* W1   = (const float*)d_in[3];
    const float* U1   = (const float*)d_in[4];
    const float* b1   = (const float*)d_in[5];
    const float* W2   = (const float*)d_in[6];
    const float* U2   = (const float*)d_in[7];
    const float* b2   = (const float*)d_in[8];
    const float* Wlm  = (const float*)d_in[9];
    const float* blm  = (const float*)d_in[10];
    const float* gam  = (const float*)d_in[11];
    const float* bet  = (const float*)d_in[12];
    const float* mea  = (const float*)d_in[13];
    const float* var  = (const float*)d_in[14];
    const float* W3   = (const float*)d_in[15];
    const float* b3   = (const float*)d_in[16];
    const float* W4   = (const float*)d_in[17];
    const float* b4   = (const float*)d_in[18];

    char* ws = (char*)d_ws;
    _Float16* h1 = (_Float16*)ws;                                        // [T][B][64] f16 = 64 MB
    float*    h2 = (float*)(ws + (size_t)T_LEN * B_TOT * DH * sizeof(_Float16));  // [B][64] f32

    lstm_layer_kernel<<<B_TOT / MT, 128, 0, stream>>>(
        W1, U1, b1, code, emb, (const _Float16*)nullptr, h1, (float*)nullptr, 1);
    lstm_layer_kernel<<<B_TOT / MT, 128, 0, stream>>>(
        W2, U2, b2, (const int*)nullptr, (const float*)nullptr, h1, (_Float16*)nullptr, h2, 2);
    head_kernel<<<(B_TOT + 127) / 128, 128, 0, stream>>>(
        h2, aux, Wlm, blm, gam, bet, mea, var, W3, b3, W4, b4, (float*)d_out);
}
